// WindowAttention_20109036880671
// MI455X (gfx1250) — matrix operand order repaired and checked by reading, not compiled, not run on hardware
//
#include <hip/hip_runtime.h>

#define WIN   7
#define NTOK  49
#define HEADS 8
#define DIM   256
#define HD    32
#define NPAD  64      // tokens padded to 64 rows
#define XS    264     // x / out LDS row stride (f16 elems), 16B-aligned, bank-friendly
#define QKS   40      // q/k LDS row stride
#define VTS   72      // vT LDS row stride (cols = tokens)
#define ATS   72      // attn LDS row stride

typedef __attribute__((ext_vector_type(16))) _Float16 v16h;
typedef __attribute__((ext_vector_type(8)))  _Float16 v8h;
typedef __attribute__((ext_vector_type(4)))  _Float16 v4h;
typedef __attribute__((ext_vector_type(8)))  float    v8f;
typedef __attribute__((ext_vector_type(4)))  float    v4f;

union Frag { v16h v; v8h h[2]; };

// A/B fragment gather from LDS: two 16-byte contiguous chunks per lane.
__device__ __forceinline__ v16h lds_frag(const _Float16* lo, const _Float16* hi) {
  Frag u;
  u.h[0] = *(const v8h*)lo;
  u.h[1] = *(const v8h*)hi;
  return u.v;
}

__device__ __forceinline__ v8f wmma_f16(v16h a, v16h b, v8f c) {
  // D = A x B + C ; f32 accumulate, emits v_wmma_f32_16x16x32_f16
  return __builtin_amdgcn_wmma_f32_16x16x32_f16(false, a, false, b, (short)0, c,
                                                false, false);
}

// ---- weight transpose + fp32->f16 convert (run once per launch, L2 resident) ----
__global__ void wa_cvt_wqkvT(const float* __restrict__ w, _Float16* __restrict__ wT) {
  int i = blockIdx.x * 256 + threadIdx.x;   // 768*256 elements
  int n = i >> 8;                            // output column of w (row of wT)
  int k = i & 255;                           // input channel
  wT[i] = (_Float16)w[k * 768 + n];
}

__global__ void wa_cvt_wprojT(const float* __restrict__ w, _Float16* __restrict__ wT) {
  int i = blockIdx.x * 256 + threadIdx.x;   // 256*256 elements
  int n = i >> 8;
  int k = i & 255;
  wT[i] = (_Float16)w[k * 256 + n];
}

// ---- fused window attention: 1 workgroup per window, 1 wave per head ----
// waves_per_eu(2): guarantee 8 waves (2/SIMD32) of this workgroup fit the VGPR file.
__global__ __launch_bounds__(256)
__attribute__((amdgpu_waves_per_eu(2)))
void wa_fused_kernel(const float* __restrict__ x,
                     const _Float16* __restrict__ wqkvT,
                     const float* __restrict__ b_qkv,
                     const _Float16* __restrict__ wprojT,
                     const float* __restrict__ b_proj,
                     const float* __restrict__ bias_table,
                     float* __restrict__ out)
{
  __shared__ _Float16 s_x [NPAD * XS];             // x window, f16
  __shared__ _Float16 s_o [NPAD * XS];             // concat head outputs
  __shared__ _Float16 s_q [HEADS][NPAD * QKS];     // q rows [token][hd], pre-scaled
  __shared__ _Float16 s_k [HEADS][NPAD * QKS];     // k rows [token][hd]
  __shared__ _Float16 s_vt[HEADS][HD   * VTS];     // v transposed [hd][token]
  __shared__ _Float16 s_at[HEADS][NPAD * ATS];     // softmax(attn) rows [q][k]

  const int tid  = threadIdx.x;
  const int lane = tid & 31;
  const int h    = tid >> 5;        // wave id == head id
  const int half = lane >> 4;       // lane half (K-offset select per ISA layout)
  const int l16  = lane & 15;
  const int b    = blockIdx.x;
  const float scale = 0.17677669529663687f;   // 32^-0.5

  const v8f vzero = {0.f, 0.f, 0.f, 0.f, 0.f, 0.f, 0.f, 0.f};

  // ---------- stage x window into LDS (f16), zero-pad rows [49,64) ----------
  // Non-temporal: each x element is read exactly once device-wide.
  const float* xb = x + (size_t)b * NTOK * DIM;
  for (int i = tid; i < NPAD * (DIM / 4); i += 256) {
    int r  = i >> 6;          // 64 float4 per row
    int c4 = (i & 63) * 4;
    v4f v = {0.f, 0.f, 0.f, 0.f};
    if (r < NTOK)
      v = __builtin_nontemporal_load(((const v4f*)xb) + r * (DIM / 4) + (i & 63));
    v4h p;
    p[0] = (_Float16)v.x; p[1] = (_Float16)v.y;
    p[2] = (_Float16)v.z; p[3] = (_Float16)v.w;
    *(v4h*)&s_x[r * XS + c4] = p;
  }
  __syncthreads();

  // ---------- QKV projection for this wave's head ----------
  // wqkvT row = s*256 + h*32 + n  (n = hd index), 256 contiguous k per row
  for (int s = 0; s < 3; ++s) {
    v8f acc[4][2];
    #pragma unroll
    for (int mt = 0; mt < 4; ++mt)
      #pragma unroll
      for (int nt = 0; nt < 2; ++nt) acc[mt][nt] = vzero;

    #pragma unroll
    for (int kk = 0; kk < 8; ++kk) {
      const int k0 = kk * 32;
      v16h af[4];
      #pragma unroll
      for (int mt = 0; mt < 4; ++mt) {
        const _Float16* p = &s_x[(mt * 16 + l16) * XS + k0 + half * 16];
        af[mt] = lds_frag(p, p + 8);
      }
      #pragma unroll
      for (int nt = 0; nt < 2; ++nt) {
        const int row = s * 256 + h * 32 + nt * 16 + l16;
        v16h bf = *(const v16h*)(wqkvT + row * 256 + k0 + half * 16);
        #pragma unroll
        for (int mt = 0; mt < 4; ++mt)
          acc[mt][nt] = wmma_f16(af[mt], bf, acc[mt][nt]);
      }
    }

    // add b_qkv, scale q, scatter into per-head LDS buffers
    #pragma unroll
    for (int nt = 0; nt < 2; ++nt) {
      const int col = nt * 16 + l16;                    // hd index
      const float bv = b_qkv[s * 256 + h * 32 + col];
      #pragma unroll
      for (int mt = 0; mt < 4; ++mt) {
        if (s == 2) {
          v8h pk;                                       // v -> vT[hd][token], contiguous
          #pragma unroll
          for (int j = 0; j < 8; ++j) pk[j] = (_Float16)(acc[mt][nt][j] + bv);
          *(v8h*)&s_vt[h][col * VTS + mt * 16 + half * 8] = pk;
        } else {
          _Float16* dst = (s == 0) ? s_q[h] : s_k[h];
          const float mul = (s == 0) ? scale : 1.0f;
          #pragma unroll
          for (int j = 0; j < 8; ++j) {
            const int m = mt * 16 + half * 8 + j;
            dst[m * QKS + col] = (_Float16)((acc[mt][nt][j] + bv) * mul);
          }
        }
      }
    }
  }

  // ---------- attention logits: att = (q*scale) @ k^T  (K=32 -> 1 wmma/tile) ----------
  v8f att[4][4];
  {
    v16h aq[4];
    #pragma unroll
    for (int mt = 0; mt < 4; ++mt) {
      const _Float16* p = &s_q[h][(mt * 16 + l16) * QKS + half * 16];
      aq[mt] = lds_frag(p, p + 8);
    }
    #pragma unroll
    for (int nt = 0; nt < 4; ++nt) {
      // B = k^T: lane column = key token, 16 contiguous hd values (half selects +16)
      const _Float16* p = &s_k[h][(nt * 16 + l16) * QKS + half * 16];
      v16h bk = lds_frag(p, p + 8);
      #pragma unroll
      for (int mt = 0; mt < 4; ++mt)
        att[mt][nt] = wmma_f16(aq[mt], bk, vzero);
    }
  }

  // ---------- relative position bias + padding mask ----------
  #pragma unroll
  for (int nt = 0; nt < 4; ++nt) {
    const int n  = nt * 16 + l16;                 // key token
    const int nh = n / WIN, nw = n % WIN;
    #pragma unroll
    for (int mt = 0; mt < 4; ++mt) {
      #pragma unroll
      for (int j = 0; j < 8; ++j) {
        const int m = mt * 16 + half * 8 + j;     // query token
        float v = att[mt][nt][j];
        if (m < NTOK && n < NTOK) {
          const int mh  = m / WIN, mw = m % WIN;
          const int rel = (mh - nh + WIN - 1) * (2 * WIN - 1) + (mw - nw + WIN - 1);
          v += bias_table[rel * HEADS + h];
        } else {
          v = -1.0e30f;
        }
        att[mt][nt][j] = v;
      }
    }
  }

  // ---------- softmax over keys (row = fixed lane-half + VGPR j, cols across lanes) ----------
  #pragma unroll
  for (int mt = 0; mt < 4; ++mt) {
    #pragma unroll
    for (int j = 0; j < 8; ++j) {
      float mx = att[mt][0][j];
      #pragma unroll
      for (int nt = 1; nt < 4; ++nt) mx = fmaxf(mx, att[mt][nt][j]);
      #pragma unroll
      for (int d = 1; d < 16; d <<= 1) mx = fmaxf(mx, __shfl_xor(mx, d, 32));
      float sum = 0.0f;
      #pragma unroll
      for (int nt = 0; nt < 4; ++nt) {
        const float e = __expf(att[mt][nt][j] - mx);
        att[mt][nt][j] = e;
        sum += e;
      }
      #pragma unroll
      for (int d = 1; d < 16; d <<= 1) sum += __shfl_xor(sum, d, 32);
      const float r = 1.0f / sum;
      #pragma unroll
      for (int nt = 0; nt < 4; ++nt) att[mt][nt][j] *= r;
    }
  }

  // spill softmax(attn) to LDS row-major [query][key] as f16
  #pragma unroll
  for (int nt = 0; nt < 4; ++nt) {
    const int n = nt * 16 + l16;
    #pragma unroll
    for (int mt = 0; mt < 4; ++mt)
      #pragma unroll
      for (int j = 0; j < 8; ++j)
        s_at[h][(mt * 16 + half * 8 + j) * ATS + n] = (_Float16)att[mt][nt][j];
  }

  // ---------- out_h = attn @ v ----------
  v8f oacc[4][2];
  #pragma unroll
  for (int mt = 0; mt < 4; ++mt)
    #pragma unroll
    for (int nt = 0; nt < 2; ++nt) oacc[mt][nt] = vzero;

  #pragma unroll
  for (int kk = 0; kk < 2; ++kk) {
    v16h aa[4];
    #pragma unroll
    for (int mt = 0; mt < 4; ++mt) {
      const _Float16* p = &s_at[h][(mt * 16 + l16) * ATS + kk * 32 + half * 16];
      aa[mt] = lds_frag(p, p + 8);
    }
    #pragma unroll
    for (int nt = 0; nt < 2; ++nt) {
      // B = v: lane column = hd index, 16 contiguous tokens from vT
      const _Float16* p = &s_vt[h][(nt * 16 + l16) * VTS + kk * 32 + half * 16];
      v16h bv = lds_frag(p, p + 8);
      #pragma unroll
      for (int mt = 0; mt < 4; ++mt)
        oacc[mt][nt] = wmma_f16(aa[mt], bv, oacc[mt][nt]);
    }
  }

  // write head output into shared concat buffer at columns [h*32, h*32+32)
  #pragma unroll
  for (int nt = 0; nt < 2; ++nt) {
    const int col = h * 32 + nt * 16 + l16;
    #pragma unroll
    for (int mt = 0; mt < 4; ++mt)
      #pragma unroll
      for (int j = 0; j < 8; ++j)
        s_o[(mt * 16 + half * 8 + j) * XS + col] = (_Float16)oacc[mt][nt][j];
  }
  __syncthreads();

  // ---------- final projection: out = concat @ w_proj + b_proj ----------
  // each wave owns 2 of 16 column tiles
  v8f pacc[4][2];
  #pragma unroll
  for (int mt = 0; mt < 4; ++mt)
    #pragma unroll
    for (int nt = 0; nt < 2; ++nt) pacc[mt][nt] = vzero;

  #pragma unroll
  for (int kk = 0; kk < 8; ++kk) {
    const int k0 = kk * 32;
    v16h ao[4];
    #pragma unroll
    for (int mt = 0; mt < 4; ++mt) {
      const _Float16* p = &s_o[(mt * 16 + l16) * XS + k0 + half * 16];
      ao[mt] = lds_frag(p, p + 8);
    }
    #pragma unroll
    for (int nt = 0; nt < 2; ++nt) {
      const int row = (h * 2 + nt) * 16 + l16;
      v16h bw = *(const v16h*)(wprojT + row * 256 + k0 + half * 16);
      #pragma unroll
      for (int mt = 0; mt < 4; ++mt)
        pacc[mt][nt] = wmma_f16(ao[mt], bw, pacc[mt][nt]);
    }
  }

  // Non-temporal stores: out written once, keep it out of L2.
  float* ob = out + (size_t)b * NTOK * DIM;
  #pragma unroll
  for (int nt = 0; nt < 2; ++nt) {
    const int col = (h * 2 + nt) * 16 + l16;
    const float bp = b_proj[col];
    #pragma unroll
    for (int mt = 0; mt < 4; ++mt)
      #pragma unroll
      for (int j = 0; j < 8; ++j) {
        const int m = mt * 16 + half * 8 + j;
        if (m < NTOK)
          __builtin_nontemporal_store(pacc[mt][nt][j] + bp, ob + m * DIM + col);
      }
  }
}

extern "C" void kernel_launch(void* const* d_in, const int* in_sizes, int n_in,
                              void* d_out, int out_size, void* d_ws, size_t ws_size,
                              hipStream_t stream) {
  (void)in_sizes; (void)n_in; (void)out_size; (void)ws_size;
  const float* x          = (const float*)d_in[0];  // [4096,49,256]
  const float* w_qkv      = (const float*)d_in[1];  // [256,768]
  const float* b_qkv      = (const float*)d_in[2];  // [768]
  const float* w_proj     = (const float*)d_in[3];  // [256,256]
  const float* b_proj     = (const float*)d_in[4];  // [256]
  const float* bias_table = (const float*)d_in[5];  // [169,8]
  float* out = (float*)d_out;

  _Float16* wqkvT  = (_Float16*)d_ws;               // [768][256] f16 = 384 KB
  _Float16* wprojT = wqkvT + 768 * 256;             // [256][256] f16 = 128 KB

  wa_cvt_wqkvT <<<768, 256, 0, stream>>>(w_qkv,  wqkvT);
  wa_cvt_wprojT<<<256, 256, 0, stream>>>(w_proj, wprojT);
  wa_fused_kernel<<<4096, 256, 0, stream>>>(x, wqkvT, b_qkv, wprojT, b_proj,
                                            bias_table, out);
}
